// GAUAttention_50337016709690
// MI455X (gfx1250) — matrix operand order repaired and checked by reading, not compiled, not run on hardware
//
#include <hip/hip_runtime.h>

#define B_  2
#define S_  2048
#define D_  1024
#define H_  16
#define DH_ 64
#define M_  4096  // B_*S_

typedef __bf16 bf16_t;
typedef __bf16 v16bf __attribute__((ext_vector_type(16)));
typedef __bf16 v8bf  __attribute__((ext_vector_type(8)));
typedef float  v8f   __attribute__((ext_vector_type(8)));

// ---------------- WMMA fragment helpers (gfx1250 wave32 layouts) ----------------
// A (16x32 bf16): lane l -> row m=l%16; elements 0..7 hold K = 8*(l/16)+0..7,
// elements 8..15 hold K = 16+8*(l/16)+0..7  (ISA 7.12.2 table).
static __device__ __forceinline__ v16bf load_a_frag(const bf16_t* p) {
  v8bf lo = *(const v8bf*)(p);
  v8bf hi = *(const v8bf*)(p + 16);
  v16bf r;
#pragma unroll
  for (int i = 0; i < 8; ++i) { r[i] = lo[i]; r[i + 8] = hi[i]; }
  return r;
}
// B (32x16 bf16): lane l -> col n=l%16; elements 0..7 hold K = 8*(l/16)+0..7,
// elements 8..15 hold K = 16+8*(l/16)+0..7 (same as A). p points at K base + 16*(l/16).
static __device__ __forceinline__ v16bf load_b_frag(const bf16_t* p) {
  const bf16_t* q = p - 8 * ((threadIdx.x & 31) >> 4);
  v8bf lo = *(const v8bf*)(q);
  v8bf hi = *(const v8bf*)(q + 16);
  v16bf r;
#pragma unroll
  for (int i = 0; i < 8; ++i) { r[i] = lo[i]; r[i + 8] = hi[i]; }
  return r;
}
static __device__ __forceinline__ v8f wmma_bf16(v16bf a, v16bf b, v8f c) {
  return __builtin_amdgcn_wmma_f32_16x16x32_bf16(false, a, false, b, (short)0, c, false, false);
}

// ---------------- hs -> bf16 + per-row mean -> gating ----------------
__global__ void convert_hs_kernel(const float* __restrict__ hs, bf16_t* __restrict__ hsb,
                                  float* __restrict__ gating, const float* __restrict__ gf,
                                  const float* __restrict__ gb) {
  int row = blockIdx.x;
  int t = threadIdx.x;
  __shared__ float red[256];
  float s = 0.f;
#pragma unroll
  for (int i = t; i < D_; i += 256) {
    float v = hs[(size_t)row * D_ + i];
    hsb[(size_t)row * D_ + i] = (bf16_t)v;
    s += v;
  }
  red[t] = s;
  __syncthreads();
  for (int st = 128; st > 0; st >>= 1) {
    if (t < st) red[t] += red[t + st];
    __syncthreads();
  }
  if (t == 0) {
    float mean = red[0] * (1.0f / (float)D_);
    float x = gf[0] * mean + gb[0];
    gating[row] = 1.0f / (1.0f + __expf(-x));
  }
}

// ---------------- W [K,N] f32 -> Wt [N,K] bf16, LDS-tiled transpose ----------------
__global__ void w_transpose_kernel(const float* __restrict__ W, bf16_t* __restrict__ Wt) {
  __shared__ bf16_t tile[32][33];  // padded: no bank conflicts on transposed read
  int tx = threadIdx.x & 31;
  int ty = threadIdx.x >> 5;                  // 8 row-groups
  int k0 = blockIdx.y * 32, n0 = blockIdx.x * 32;
#pragma unroll
  for (int r = ty; r < 32; r += 8)
    tile[r][tx] = (bf16_t)W[(size_t)(k0 + r) * D_ + n0 + tx];   // coalesced read
  __syncthreads();
#pragma unroll
  for (int r = ty; r < 32; r += 8)
    Wt[(size_t)(n0 + r) * D_ + k0 + tx] = tile[tx][r];          // coalesced write
}

// ---------------- WMMA GEMM: C[M,1024] = A[M,1024] x Wt^T + bias ----------------
// Each wave computes a 32x64 tile: 2 A-frags x 4 B-frags -> 8 WMMAs per k-step,
// so 6 vector loads feed 8 WMMAs (vs 2 loads per WMMA in a naive tiling).
// out modes: 0 = bf16 [B,H,S,DH] ; 1 = bf16 [B,H,DH,S] (transposed V) ; 2 = f32 [M,N]
__global__ void gemm_bf16_wmma(const bf16_t* __restrict__ A, const bf16_t* __restrict__ Wt,
                               const float* __restrict__ bias, void* __restrict__ out,
                               int mode) {
  const int K = D_, N = D_;
  const int nBlk = N / 256;                    // 4 wave-column-groups of 64
  int mtile = (blockIdx.x / nBlk) * 32;
  int wave  = threadIdx.x >> 5;
  int lane  = threadIdx.x & 31;
  int ntile = (blockIdx.x % nBlk) * 256 + wave * 64;
  int lm = lane & 15, hi = lane >> 4;

  const bf16_t* ap0 = A + (size_t)(mtile + lm) * K + hi * 8;
  const bf16_t* ap1 = ap0 + (size_t)16 * K;
  const bf16_t* bp  = Wt + (size_t)(ntile + lm) * K + hi * 16;

  v8f acc[2][4];
#pragma unroll
  for (int i = 0; i < 2; ++i)
#pragma unroll
    for (int j = 0; j < 4; ++j) acc[i][j] = (v8f){};

  for (int k0 = 0; k0 < K; k0 += 32) {
    if (k0 + 512 < K) {                        // cover streaming latency
      __builtin_prefetch(ap0 + k0 + 512, 0, 1);
      __builtin_prefetch(bp + k0 + 512, 0, 1);
    }
    v16bf a0 = load_a_frag(ap0 + k0);
    v16bf a1 = load_a_frag(ap1 + k0);
#pragma unroll
    for (int j = 0; j < 4; ++j) {
      v16bf b = load_b_frag(bp + (size_t)j * 16 * K + k0);
      acc[0][j] = wmma_bf16(a0, b, acc[0][j]);
      acc[1][j] = wmma_bf16(a1, b, acc[1][j]);
    }
  }

#pragma unroll
  for (int i = 0; i < 2; ++i) {
    int mt = mtile + 16 * i;
#pragma unroll
    for (int j = 0; j < 4; ++j) {
      int col = ntile + j * 16 + lm;
      float bval = bias[col];
      v8f c = acc[i][j];
      if (mode == 2) {
        float* o = (float*)out;
#pragma unroll
        for (int r = 0; r < 8; ++r)
          o[(size_t)(mt + r + 8 * hi) * N + col] = c[r] + bval;
      } else if (mode == 0) {
        bf16_t* o = (bf16_t*)out;
        int b = mt >> 11;
        int h = col >> 6, dh = col & 63;
#pragma unroll
        for (int r = 0; r < 8; ++r) {
          int s = (mt & (S_ - 1)) + r + 8 * hi;
          o[((size_t)(b * H_ + h) * S_ + s) * DH_ + dh] = (bf16_t)(c[r] + bval);
        }
      } else {  // mode 1: V transposed; rows r contiguous in S -> one packed store
        bf16_t* o = (bf16_t*)out;
        int b = mt >> 11;
        int h = col >> 6, dh = col & 63;
        int s0 = (mt & (S_ - 1)) + 8 * hi;
        v8bf pack;
#pragma unroll
        for (int r = 0; r < 8; ++r) pack[r] = (bf16_t)(c[r] + bval);
        *(v8bf*)(o + ((size_t)(b * H_ + h) * DH_ + dh) * S_ + s0) = pack;
      }
    }
  }
}

// ---------------- Flash attention: single-wave workgroups (barrier -> S_NOP) ----
// Q,K: bf16 [B,H,S,DH]; Vt: bf16 [B,H,DH,S]; ctx out: bf16 [B*S, D] row-major.
__global__ void attn_wmma(const bf16_t* __restrict__ Q, const bf16_t* __restrict__ Kc,
                          const bf16_t* __restrict__ Vt, const float* __restrict__ gating,
                          bf16_t* __restrict__ ctx) {
  const int qTiles = S_ / 16;                   // 128 query tiles per (b,h)
  int bh = blockIdx.x / qTiles;
  int b = bh >> 4, h = bh & 15;
  int lane = threadIdx.x & 31;
  int q0 = (blockIdx.x % qTiles) * 16;
  int lm = lane & 15, hi = lane >> 4;

  const bf16_t* qbase = Q  + (size_t)bh * S_ * DH_;
  const bf16_t* kbase = Kc + (size_t)bh * S_ * DH_;
  const bf16_t* vbase = Vt + (size_t)bh * DH_ * S_;

  v16bf aq0 = load_a_frag(qbase + (size_t)(q0 + lm) * DH_ + 0  + hi * 8);
  v16bf aq1 = load_a_frag(qbase + (size_t)(q0 + lm) * DH_ + 32 + hi * 8);

  float runM[8], runL[8];
  v8f o0 = {}, o1 = {}, o2 = {}, o3 = {};
#pragma unroll
  for (int r = 0; r < 8; ++r) { runM[r] = -1e30f; runL[r] = 0.f; }

  __shared__ bf16_t pb[16 * 32];                // per-wave P transpose tile (1 KB)

  for (int j0 = 0; j0 < S_; j0 += 32) {
    const bf16_t* kp0 = kbase + (size_t)(j0 + lm) * DH_ + hi * 16;
    const bf16_t* kp1 = kbase + (size_t)(j0 + 16 + lm) * DH_ + hi * 16;
    const bf16_t* vp  = vbase + (size_t)lm * S_ + j0 + hi * 16;
    if (j0 + 64 < S_) {                         // prefetch next key/value block
      __builtin_prefetch(kp0 + 64 * DH_, 0, 1);
      __builtin_prefetch(vp + 64, 0, 1);
    }
    // ---- S = Q K^T for 32 keys (two 16-col fragments), scaled by 1/sqrt(64)
    v8f sl = {}, sh = {};
    sl = wmma_bf16(aq0, load_b_frag(kp0), sl);
    sl = wmma_bf16(aq1, load_b_frag(kp0 + 32), sl);
    sh = wmma_bf16(aq0, load_b_frag(kp1), sh);
    sh = wmma_bf16(aq1, load_b_frag(kp1 + 32), sh);

    float bm[8], sf[8], bs[8];
#pragma unroll
    for (int r = 0; r < 8; ++r) {
      sl[r] *= 0.125f; sh[r] *= 0.125f;
      bm[r] = fmaxf(sl[r], sh[r]);
    }
    // row max across the 16 lanes of each half (rows live in (vgpr r, half hi))
#pragma unroll
    for (int r = 0; r < 8; ++r) {
      bm[r] = fmaxf(bm[r], __shfl_xor(bm[r], 1, 32));
      bm[r] = fmaxf(bm[r], __shfl_xor(bm[r], 2, 32));
      bm[r] = fmaxf(bm[r], __shfl_xor(bm[r], 4, 32));
      bm[r] = fmaxf(bm[r], __shfl_xor(bm[r], 8, 32));
    }
#pragma unroll
    for (int r = 0; r < 8; ++r) {
      float nm = fmaxf(runM[r], bm[r]);
      sf[r] = __expf(runM[r] - nm);
      runM[r] = nm;
      float pl = __expf(sl[r] - nm), ph = __expf(sh[r] - nm);
      sl[r] = pl; sh[r] = ph; bs[r] = pl + ph;
    }
#pragma unroll
    for (int r = 0; r < 8; ++r) {
      bs[r] += __shfl_xor(bs[r], 1, 32);
      bs[r] += __shfl_xor(bs[r], 2, 32);
      bs[r] += __shfl_xor(bs[r], 4, 32);
      bs[r] += __shfl_xor(bs[r], 8, 32);
      runL[r] = runL[r] * sf[r] + bs[r];
      o0[r] *= sf[r]; o1[r] *= sf[r]; o2[r] *= sf[r]; o3[r] *= sf[r];
    }
    // ---- transpose P (C-layout) into A-layout via LDS tile (wave-local;
    //      single-wave workgroup -> barrier lowers to S_NOP, LDS is in-order)
#pragma unroll
    for (int r = 0; r < 8; ++r) {
      int m = r + 8 * hi;
      pb[m * 32 + lm]      = (bf16_t)sl[r];
      pb[m * 32 + 16 + lm] = (bf16_t)sh[r];
    }
    __syncthreads();
    v16bf apfrag = load_a_frag(pb + lm * 32 + hi * 8);
    // ---- O += P V  (V transposed so B-frags are contiguous 16B loads)
    o0 = wmma_bf16(apfrag, load_b_frag(vp), o0);
    o1 = wmma_bf16(apfrag, load_b_frag(vp + 16 * S_), o1);
    o2 = wmma_bf16(apfrag, load_b_frag(vp + 32 * S_), o2);
    o3 = wmma_bf16(apfrag, load_b_frag(vp + 48 * S_), o3);
    __syncthreads();
  }

  // normalize, apply gating, store ctx [B*S, D] bf16
#pragma unroll
  for (int r = 0; r < 8; ++r) {
    int qrow = q0 + r + 8 * hi;
    float g = gating[b * S_ + qrow];
    float inv = g / runL[r];
    size_t rowoff = ((size_t)(b * S_ + qrow)) * D_ + h * DH_ + lm;
    ctx[rowoff + 0]  = (bf16_t)(o0[r] * inv);
    ctx[rowoff + 16] = (bf16_t)(o1[r] * inv);
    ctx[rowoff + 32] = (bf16_t)(o2[r] * inv);
    ctx[rowoff + 48] = (bf16_t)(o3[r] * inv);
  }
}

// ---------------- host launch ----------------
extern "C" void kernel_launch(void* const* d_in, const int* in_sizes, int n_in,
                              void* d_out, int out_size, void* d_ws, size_t ws_size,
                              hipStream_t stream) {
  (void)in_sizes; (void)n_in; (void)out_size; (void)ws_size;
  const float* hs = (const float*)d_in[0];
  const float* Wq = (const float*)d_in[1];
  const float* bq = (const float*)d_in[2];
  const float* Wk = (const float*)d_in[3];
  const float* bk = (const float*)d_in[4];
  const float* Wv = (const float*)d_in[5];
  const float* bv = (const float*)d_in[6];
  const float* Wo = (const float*)d_in[7];
  const float* bo = (const float*)d_in[8];
  const float* gf = (const float*)d_in[9];
  const float* gb = (const float*)d_in[10];

  char* ws = (char*)d_ws;
  size_t off = 0;
  bf16_t* hsb  = (bf16_t*)(ws + off); off += (size_t)M_ * D_ * 2;
  bf16_t* wqt  = (bf16_t*)(ws + off); off += (size_t)D_ * D_ * 2;
  bf16_t* wkt  = (bf16_t*)(ws + off); off += (size_t)D_ * D_ * 2;
  bf16_t* wvt  = (bf16_t*)(ws + off); off += (size_t)D_ * D_ * 2;
  bf16_t* wot  = (bf16_t*)(ws + off); off += (size_t)D_ * D_ * 2;
  bf16_t* qb   = (bf16_t*)(ws + off); off += (size_t)M_ * D_ * 2;
  bf16_t* kbuf = (bf16_t*)(ws + off); off += (size_t)M_ * D_ * 2;
  bf16_t* vtb  = (bf16_t*)(ws + off); off += (size_t)M_ * D_ * 2;
  bf16_t* ctxb = (bf16_t*)(ws + off); off += (size_t)M_ * D_ * 2;
  float*  gating = (float*)(ws + off); off += (size_t)M_ * 4;

  convert_hs_kernel<<<M_, 256, 0, stream>>>(hs, hsb, gating, gf, gb);
  dim3 tgrid(D_ / 32, D_ / 32);
  w_transpose_kernel<<<tgrid, 256, 0, stream>>>(Wq, wqt);
  w_transpose_kernel<<<tgrid, 256, 0, stream>>>(Wk, wkt);
  w_transpose_kernel<<<tgrid, 256, 0, stream>>>(Wv, wvt);
  w_transpose_kernel<<<tgrid, 256, 0, stream>>>(Wo, wot);

  dim3 ggrid((M_ / 32) * (D_ / 256));  // 512 blocks, 4 waves, 32x64 per wave
  gemm_bf16_wmma<<<ggrid, 128, 0, stream>>>(hsb, wqt, bq, qb,   0);
  gemm_bf16_wmma<<<ggrid, 128, 0, stream>>>(hsb, wkt, bk, kbuf, 0);
  gemm_bf16_wmma<<<ggrid, 128, 0, stream>>>(hsb, wvt, bv, vtb,  1);

  attn_wmma<<<B_ * H_ * (S_ / 16), 32, 0, stream>>>(qb, kbuf, vtb, gating, ctxb);

  gemm_bf16_wmma<<<ggrid, 128, 0, stream>>>(ctxb, wot, bo, d_out, 2);
}
